// Model_2680059593261
// MI455X (gfx1250) — matrix operand order repaired and checked by reading, not compiled, not run on hardware
//
#include <hip/hip_runtime.h>
#include <hip/hip_bf16.h>
#include <math.h>

// Problem constants
#define BB   256
#define NN   128
#define HH   512
#define EE   2048
#define OUTD 32
#define KMAX 128      // max distinct degree classes (<= NN)

typedef __attribute__((ext_vector_type(16))) __bf16         v16bf;
typedef __attribute__((ext_vector_type(8)))  float          v8f;
typedef __attribute__((ext_vector_type(8)))  unsigned short ushort8;
typedef __attribute__((ext_vector_type(4)))  unsigned short usvec4;

static __device__ __forceinline__ unsigned short f2bf(float f) {
    unsigned int u = __float_as_uint(f);
    unsigned int r = u + 0x7FFFu + ((u >> 16) & 1u);   // round-to-nearest-even
    return (unsigned short)(r >> 16);
}
static __device__ __forceinline__ float bf2f(unsigned short h) {
    return __uint_as_float(((unsigned int)h) << 16);
}

// ---------------------------------------------------------------------------
// P1: graph preprocessing (batch-independent, runs once per launch).
//   deg(n) -> distinct coefficient classes c_j = 1+degval_j  (first-seen order)
//   CNT'[n][j] = #incoming edges from class-j sources + (class(n)==j)
// Then z[b] = CNT' @ G[b] with G[b][j,:] = relu(c_j*S1[b,:] + b1).
// CNT' is written pre-swizzled in bf16 WMMA A-operand layout:
//   tile (mt,kt): lane l holds row M=mt*16+(l&15),
//   K = kt*32 + ((l>>4)<<3) + (j<8 ? j : j+8), 32B contiguous per lane.
// ---------------------------------------------------------------------------
__global__ void build_graph(const int* __restrict__ edge_index,
                            float* __restrict__ dvals_g,   // 128 f32 (padded 0)
                            int*   __restrict__ nkt_g,     // #32-wide K steps
                            unsigned short* __restrict__ Acnt) { // 8*4*32*16 bf16
    __shared__ int deg[NN];
    __shared__ int cls[NN];
    __shared__ float dv[NN];
    __shared__ int Dsh;
    __shared__ unsigned short CNT[NN][KMAX];   // 32 KB
    int t = threadIdx.x;
    if (t < NN) deg[t] = 0;
    for (int i = t; i < NN * KMAX; i += 256) ((unsigned short*)CNT)[i] = 0;
    __syncthreads();
    for (int e = t; e < EE; e += 256) atomicAdd(&deg[edge_index[EE + e]], 1);
    __syncthreads();
    if (t == 0) {
        int D = 0;
        for (int n = 0; n < NN; n++) {               // first-seen distinct degrees
            int c = -1;
            for (int j = 0; j < D; j++)
                if ((int)dv[j] == 1 + deg[n]) { c = j; break; }
            if (c < 0) { c = D; dv[D++] = (float)(1 + deg[n]); }
            cls[n] = c;
        }
        Dsh = D;
        for (int e = 0; e < EE; e++)                 // deterministic serial fill
            CNT[edge_index[EE + e]][cls[edge_index[e]]]++;
        for (int n = 0; n < NN; n++) CNT[n][cls[n]]++;   // self term (x + agg)
    }
    __syncthreads();
    if (t < NN) dvals_g[t] = (t < Dsh) ? dv[t] : 0.f;
    if (t == 0) nkt_g[0] = (Dsh + 31) >> 5;
    // swizzle CNT' into A-operand bf16 layout (counts are exact in bf16)
    for (int idx = t; idx < 8 * 4 * 32 * 16; idx += 256) {
        int j    = idx & 15;
        int lane = (idx >> 4) & 31;
        int kt   = (idx >> 9) & 3;
        int mt   = idx >> 11;
        int M = mt * 16 + (lane & 15);
        int K = kt * 32 + ((lane >> 4) << 3) + (j < 8 ? j : j + 8);
        Acnt[idx] = f2bf((float)CNT[M][K]);
    }
}

// ---------------------------------------------------------------------------
// P2: W2 (512x512 f32 row-major [K][N]) -> bf16 in WMMA B-operand layout.
//   tile (kt,nt): lane l holds column N=nt*16+(l&15), K=kt*32+(l>>4)*8+(j<8?j:j+8),
//   stored at W2s[(((kt*32+nt)*32)+lane)*16 + j] -> 32B contiguous per lane.
// ---------------------------------------------------------------------------
__global__ void swizzle_w2(const float* __restrict__ W2,
                           unsigned short* __restrict__ W2s) {
    int idx  = blockIdx.x * 256 + threadIdx.x;
    int j    = idx & 15;
    int lane = (idx >> 4) & 31;
    int nt   = (idx >> 9) & 31;
    int kt   = idx >> 14;
    int N    = nt * 16 + (lane & 15);
    int K    = kt * 32 + ((lane >> 4) << 3) + (j < 8 ? j : j + 8);
    W2s[idx] = f2bf(W2[K * HH + N]);
}

// ---------------------------------------------------------------------------
// Fused per-batch kernel: one workgroup (8 wave32) per batch element.
// ---------------------------------------------------------------------------
__global__ __launch_bounds__(256) void fused_kernel(
        const float* __restrict__ state, const float* __restrict__ W1,
        const float* __restrict__ b1,    const float* __restrict__ b2,
        const float* __restrict__ Wout,  const float* __restrict__ bout,
        const float* __restrict__ dvals_g, const int* __restrict__ nkt_g,
        const unsigned short* __restrict__ Acnt,
        const unsigned short* __restrict__ W2s,
        float* __restrict__ out) {
    extern __shared__ char smem[];
    unsigned short* zbf = (unsigned short*)smem;      // 128x512 bf16 (z)
    unsigned short* gt  = zbf + NN * HH;              // Gt[512][128] bf16, later h2[128][512]
    unsigned short* hbf = gt;                         // alias after barrier
    float* S1     = (float*)(gt + NN * HH);           // 512
    float* b1s    = S1   + HH;                        // 512
    float* b2s    = b1s  + HH;                        // 512
    float* sv     = b2s  + HH;                        // 128
    float* dvs    = sv   + NN;                        // 128
    float* outacc = dvs  + NN;                        // 8*4*32 = 1024

    const int b = blockIdx.x;
    const int t = threadIdx.x;
    const int wave = t >> 5;
    const int lane = t & 31;
    const int lrow16 = lane & 15;
    const int khalf8 = (lane >> 4) << 3;
    const int nkt = nkt_g[0];                         // uniform (1..4)

    // ---- stage 0: stage small data into LDS -------------------------------
    if (t < NN) { sv[t] = state[b * NN + t]; dvs[t] = dvals_g[t]; }
    for (int i = t; i < HH; i += 256) { b1s[i] = b1[i]; b2s[i] = b2[i]; }
    __syncthreads();

    // ---- stage 1: S1 = state[b] @ W1 (1x128 @ 128x512) --------------------
    for (int h = t; h < HH; h += 256) {
        float acc = 0.f;
        #pragma unroll 8
        for (int f = 0; f < NN; f++) acc = fmaf(sv[f], W1[f * HH + h], acc);
        S1[h] = acc;
    }
    __syncthreads();

    // ---- stage 2: Gt[f][k] = relu(c_k*S1[f] + b1[f])  (B operand) ---------
    {
        const int K32 = nkt << 5;
        for (int f = t; f < HH; f += 256) {
            float s1 = S1[f], bb = b1s[f];
            for (int k = 0; k < K32; k++) {
                float v = fmaf(dvs[k], s1, bb);
                gt[f * KMAX + k] = f2bf(v > 0.f ? v : 0.f);
            }
        }
    }
    __syncthreads();

    // ---- stage 3: z = CNT' @ G  via WMMA (graph aggregation as GEMM) ------
    for (int i = 0; i < 32; i++) {
        int tile = wave + (i << 3);
        int mt = tile >> 5, nt = tile & 31;
        v8f acc = {0.f, 0.f, 0.f, 0.f, 0.f, 0.f, 0.f, 0.f};
        for (int kt = 0; kt < nkt; kt++) {
            union { v16bf v; ushort8 u[2]; } A, Bv;
            const ushort8* ap =
                (const ushort8*)&Acnt[((((mt << 2) | kt) << 5) | lane) << 4];
            A.u[0] = ap[0];
            A.u[1] = ap[1];
            int gi = (nt * 16 + lrow16) * KMAX + (kt << 5) + ((lane >> 4) << 3);
            Bv.u[0] = *(const ushort8*)&gt[gi];
            Bv.u[1] = *(const ushort8*)&gt[gi + 16];
            acc = __builtin_amdgcn_wmma_f32_16x16x32_bf16(
                      false, A.v, false, Bv.v, (short)0, acc, false, false);
        }
        int ncol  = nt * 16 + lrow16;
        int rbase = mt * 16 + khalf8;
        #pragma unroll
        for (int r = 0; r < 8; r++)
            zbf[(rbase + r) * HH + ncol] = f2bf(acc[r]);   // z: no bias/relu
    }
    __syncthreads();

    // ---- stage 4: h2 = relu(z @ W2 + b2) via WMMA -------------------------
    for (int i = 0; i < 32; i++) {
        int tile = wave + (i << 3);
        int mt = tile >> 5, nt = tile & 31;
        v8f acc = {0.f, 0.f, 0.f, 0.f, 0.f, 0.f, 0.f, 0.f};
        int arow = mt * 16 + lrow16;
        for (int kt = 0; kt < 16; kt++) {
            int abase = kt * 32 + khalf8;
            union { v16bf v; ushort8 u[2]; } A, Bv;
            A.u[0] = *(const ushort8*)&zbf[arow * HH + abase];       // K {0..7}
            A.u[1] = *(const ushort8*)&zbf[arow * HH + abase + 16];  // K {16..23}
            const ushort8* bp =
                (const ushort8*)&W2s[((((kt << 5) | nt) << 5) | lane) << 4];
            Bv.u[0] = bp[0];
            Bv.u[1] = bp[1];
            acc = __builtin_amdgcn_wmma_f32_16x16x32_bf16(
                      false, A.v, false, Bv.v, (short)0, acc, false, false);
        }
        int ncol  = nt * 16 + lrow16;
        float bia = b2s[ncol];
        int rbase = mt * 16 + khalf8;
        #pragma unroll
        for (int r = 0; r < 8; r++) {
            float v = acc[r] + bia;
            hbf[(rbase + r) * HH + ncol] = f2bf(v > 0.f ? v : 0.f);  // h2
        }
    }
    __syncthreads();

    // ---- stage 5: out[b,:] = tanh(flat(h2) @ Wout + bout) -----------------
    // lane -> 4 output cols (float4 Wout loads) x 4-deep flat stratum.
    {
        float a0 = 0.f, a1 = 0.f, a2 = 0.f, a3 = 0.f;
        const int s  = lane >> 3;             // flat stratum 0..3
        const int og = lane & 7;              // output group: cols og*4..og*4+3
        const float4* W4 = (const float4*)Wout;   // 8 float4 per Wout row
        const int base = wave * (NN * HH / 8) + s * 4;
        for (int it = 0; it < 512; it++) {
            int fi0 = base + it * 16;
            usvec4 hv = *(const usvec4*)&hbf[fi0];
            #pragma unroll
            for (int q = 0; q < 4; q++) {
                float hf = bf2f(hv[q]);
                float4 w = W4[(size_t)(fi0 + q) * 8 + og];
                a0 = fmaf(hf, w.x, a0);
                a1 = fmaf(hf, w.y, a1);
                a2 = fmaf(hf, w.z, a2);
                a3 = fmaf(hf, w.w, a3);
            }
        }
        int ob = wave * 128 + s * 32 + og * 4;
        outacc[ob + 0] = a0; outacc[ob + 1] = a1;
        outacc[ob + 2] = a2; outacc[ob + 3] = a3;
    }
    __syncthreads();
    if (t < OUTD) {
        float sum = 0.f;
        #pragma unroll
        for (int w = 0; w < 8; w++)
            #pragma unroll
            for (int s = 0; s < 4; s++)
                sum += outacc[w * 128 + s * 32 + t];
        out[b * OUTD + t] = tanhf(sum + bout[t]);
    }
}

// ---------------------------------------------------------------------------
extern "C" void kernel_launch(void* const* d_in, const int* in_sizes, int n_in,
                              void* d_out, int out_size, void* d_ws, size_t ws_size,
                              hipStream_t stream) {
    const float* state = (const float*)d_in[0];
    const int*   eidx  = (const int*)  d_in[1];
    const float* W1    = (const float*)d_in[2];
    const float* b1    = (const float*)d_in[3];
    const float* W2    = (const float*)d_in[4];
    const float* b2    = (const float*)d_in[5];
    const float* Wout  = (const float*)d_in[6];
    const float* bout  = (const float*)d_in[7];
    float* out = (float*)d_out;

    // workspace layout (16B-aligned sections)
    float* dvals = (float*)d_ws;                           // 128 f32
    int*   nkt   = (int*)(dvals + NN);                     // 1 (pad 4)
    unsigned short* Acnt = (unsigned short*)(nkt + 4);     // 8*4*32*16 = 32 KB
    unsigned short* W2s  = Acnt + 8 * 4 * 32 * 16;         // 512 KB

    build_graph<<<1, 256, 0, stream>>>(eidx, dvals, nkt, Acnt);
    swizzle_w2<<<(HH * HH) / 256, 256, 0, stream>>>(W2, W2s);

    constexpr size_t LDS_BYTES =
        (size_t)NN * HH * 2 * 2            // zbf + gt/h2
        + (3 * HH + 2 * NN + 1024) * 4;    // S1,b1s,b2s,sv,dvs,outacc  (~267 KB)
    (void)hipFuncSetAttribute(reinterpret_cast<const void*>(fused_kernel),
                              hipFuncAttributeMaxDynamicSharedMemorySize,
                              (int)LDS_BYTES);
    fused_kernel<<<BB, 256, LDS_BYTES, stream>>>(
        state, W1, b1, b2, Wout, bout, dvals, nkt, Acnt, W2s, out);
}
